// AtomicHAR_50508815401708
// MI455X (gfx1250) — compile-verified, matrix operand order checked by reading, not run on hardware
//
#include <hip/hip_runtime.h>

// ---------------- CDNA5 WMMA types ----------------
typedef __attribute__((ext_vector_type(8)))  _Float16 v8h;
typedef __attribute__((ext_vector_type(16))) _Float16 v16h;
typedef __attribute__((ext_vector_type(8)))  float    v8f;

#define NS   16384     // bs*seq
#define SEQ  128
#define BSZ  128
#define MAL  10
#define WTOT 1280      // MAL*SEQ
#define MAXSEG 160

// =====================================================================
// Encoder conv stack (tiny channel counts -> VALU)
// =====================================================================
__global__ void conv1_kernel(const float* __restrict__ x, const float* __restrict__ w,
                             const float* __restrict__ b, float* __restrict__ h1) {
    int tid = blockIdx.x * blockDim.x + threadIdx.x;
    if (tid >= NS * 16) return;
    int n = tid >> 4, oc = tid & 15;
    const float* xs = x + (size_t)n * 120;
    float y[20];
    float bb = b[oc];
    for (int p = 0; p < 20; ++p) y[p] = bb;
    for (int ic = 0; ic < 6; ++ic) {
        const float* wv = w + (oc * 6 + ic) * 3;
        float w0 = wv[0], w1 = wv[1], w2 = wv[2];
        const float* row = xs + ic * 20;
        for (int p = 0; p < 20; ++p) {
            float s = row[p] * w1;
            if (p > 0)  s += row[p - 1] * w0;
            if (p < 19) s += row[p + 1] * w2;
            y[p] += s;
        }
    }
    float* out = h1 + (size_t)n * 160 + oc * 10;
    for (int q = 0; q < 10; ++q) out[q] = fmaxf(fmaxf(y[2 * q], y[2 * q + 1]), 0.f);
}

__global__ void conv2_kernel(const float* __restrict__ h1, const float* __restrict__ w,
                             const float* __restrict__ b, float* __restrict__ h2) {
    int tid = blockIdx.x * blockDim.x + threadIdx.x;
    if (tid >= NS * 32) return;
    int n = tid >> 5, oc = tid & 31;
    const float* in = h1 + (size_t)n * 160;
    float y[10];
    float bb = b[oc];
    for (int p = 0; p < 10; ++p) y[p] = bb;
    for (int ic = 0; ic < 16; ++ic) {
        const float* wv = w + (oc * 16 + ic) * 3;
        float w0 = wv[0], w1 = wv[1], w2 = wv[2];
        const float* row = in + ic * 10;
        for (int p = 0; p < 10; ++p) {
            float s = row[p] * w1;
            if (p > 0) s += row[p - 1] * w0;
            if (p < 9) s += row[p + 1] * w2;
            y[p] += s;
        }
    }
    float* out = h2 + (size_t)n * 160 + oc * 5;
    for (int q = 0; q < 5; ++q) out[q] = fmaxf(fmaxf(y[2 * q], y[2 * q + 1]), 0.f);
}

__global__ void conv3_kernel(const float* __restrict__ h2, const float* __restrict__ w,
                             const float* __restrict__ b, _Float16* __restrict__ cnn) {
    int tid = blockIdx.x * blockDim.x + threadIdx.x;
    if (tid >= NS * 32) return;
    int n = tid >> 5, oc = tid & 31;
    const float* in = h2 + (size_t)n * 160;
    float y[5];
    float bb = b[oc];
    for (int p = 0; p < 5; ++p) y[p] = bb;
    for (int ic = 0; ic < 32; ++ic) {
        const float* wv = w + (oc * 32 + ic) * 3;
        float w0 = wv[0], w1 = wv[1], w2 = wv[2];
        const float* row = in + ic * 5;
        for (int p = 0; p < 5; ++p) {
            float s = row[p] * w1;
            if (p > 0) s += row[p - 1] * w0;
            if (p < 4) s += row[p + 1] * w2;
            y[p] += s;
        }
    }
    _Float16* out = cnn + (size_t)n * 64 + oc * 2;
    out[0] = (_Float16)fmaxf(fmaxf(y[0], y[1]), 0.f);
    out[1] = (_Float16)fmaxf(fmaxf(y[2], y[3]), 0.f);
}

// =====================================================================
// WMMA GEMM machinery (v_wmma_f32_16x16x32_f16)
// =====================================================================

// Pack W (K x Nout, row-major f32) into per-lane B fragments matching the
// CDNA5 16-bit 32x16 B-matrix VGPR layout (mirror of the A layout):
// lane L: n = L&15, khalf = L>>4; lane elem j<8 -> k = khalf*8+j,
// j>=8 -> k = 16 + khalf*8 + (j-8).  Fragment (nt,kt) = 512 halves.
__global__ void pack_b_kernel(const float* __restrict__ W, int K, int Nout,
                              _Float16* __restrict__ Bp) {
    int KT = K >> 5;
    size_t total = (size_t)KT * (Nout >> 4) * 512;
    for (size_t idx = blockIdx.x * (size_t)blockDim.x + threadIdx.x; idx < total;
         idx += (size_t)gridDim.x * blockDim.x) {
        int j    = (int)(idx & 15);
        int lane = (int)((idx >> 4) & 31);
        size_t t = idx >> 9;
        int kt = (int)(t % KT);
        int nt = (int)(t / KT);
        int n = lane & 15;
        int khalf = lane >> 4;
        int k = (j < 8) ? (khalf * 8 + j) : (16 + khalf * 8 + (j - 8));
        Bp[idx] = (_Float16)W[(size_t)(kt * 32 + k) * Nout + nt * 16 + n];
    }
}

__device__ __forceinline__ v16h load_a_frag(const _Float16* __restrict__ A, int lda,
                                            int row0, int kt, int lane) {
    int row   = row0 + (lane & 15);
    int kbase = kt * 32 + ((lane & 16) ? 8 : 0);
    const _Float16* p = A + (size_t)row * lda + kbase;
    v8h lo = *(const v8h*)(p);
    v8h hi = *(const v8h*)(p + 16);
    return __builtin_shufflevector(lo, hi, 0, 1, 2, 3, 4, 5, 6, 7,
                                   8, 9, 10, 11, 12, 13, 14, 15);
}

// One wave computes a 16-row M tile across all NT column tiles.
// act: 0=none 1=relu 2=sigmoid.  shiftRows: write f16 copy at row+1
// (skip t==seq-1) to build the forecast shift input.
template <int KT>
__global__ void gemm_wmma(const _Float16* __restrict__ A, int lda,
                          const _Float16* __restrict__ Bp,
                          const float* __restrict__ bias,
                          int Mtiles, int NT,
                          float* __restrict__ outF32, int ldo, int Mvalid,
                          _Float16* __restrict__ outF16, int ldh,
                          int act, int shiftRows) {
    int wave = (blockIdx.x * blockDim.x + threadIdx.x) >> 5;
    int lane = threadIdx.x & 31;
    if (wave >= Mtiles) return;   // wave-uniform: EXEC stays all-ones for WMMA
    int row0 = wave * 16;

    v16h af[KT];
#pragma unroll
    for (int kt = 0; kt < KT; ++kt) af[kt] = load_a_frag(A, lda, row0, kt, lane);

    int col     = lane & 15;
    int rowhalf = (lane >> 4) * 8;

    for (int nt = 0; nt < NT; ++nt) {
        v8f acc = {0.f, 0.f, 0.f, 0.f, 0.f, 0.f, 0.f, 0.f};
        const _Float16* bp = Bp + (size_t)nt * KT * 512 + (size_t)lane * 16;
        __builtin_prefetch(bp + (size_t)KT * 512, 0, 0);
#pragma unroll
        for (int kt = 0; kt < KT; ++kt) {
            v16h bf = *(const v16h*)(bp + (size_t)kt * 512);
            acc = __builtin_amdgcn_wmma_f32_16x16x32_f16(
                false, af[kt], false, bf, (short)0, acc, false, false);
        }
        int c0 = nt * 16 + col;
        float bv = bias ? bias[c0] : 0.f;
#pragma unroll
        for (int r = 0; r < 8; ++r) {
            int gr = row0 + rowhalf + r;
            float v = acc[r] + bv;
            if (act == 1) v = fmaxf(v, 0.f);
            else if (act == 2) v = 1.f / (1.f + __expf(-v));
            if (gr < Mvalid) {
                if (outF32) outF32[(size_t)gr * ldo + c0] = v;
                if (outF16) {
                    int hr = gr;
                    bool ok = true;
                    if (shiftRows) { ok = ((gr & (SEQ - 1)) != SEQ - 1); hr = gr + 1; }
                    if (ok) outF16[(size_t)hr * ldh + c0] = (_Float16)v;
                }
            }
        }
    }
}

// =====================================================================
// Loss / mask / scores
// =====================================================================
__global__ void loss_kernel(const float* __restrict__ forcast, const float* __restrict__ bridge,
                            const float* __restrict__ imu_mask,
                            float* __restrict__ fmask_out, float* __restrict__ loss_out,
                            float* __restrict__ scores) {
    int n = blockIdx.x * blockDim.x + threadIdx.x;
    if (n >= NS) return;
    float fm = ((n & (SEQ - 1)) != 0 ? 1.f : 0.f) * imu_mask[(size_t)n * 120];
    float s = 0.f;
    for (int j = 0; j < 64; ++j) {
        float d = (forcast[(size_t)n * 64 + j] - bridge[(size_t)n * 64 + j]) * fm;
        s += d * d;
        fmask_out[(size_t)n * 64 + j] = fm;
    }
    float l = s * (1.f / 64.f);
    loss_out[n] = l;
    scores[n]   = l * fm;
}

// =====================================================================
// Exact k-th-largest (k = 0.2*N) via 32-pass radix select on monotone keys
// =====================================================================
__global__ void select_cutoff_kernel(const float* __restrict__ scores,
                                     float* __restrict__ cutoff_out) {
    __shared__ int cnt;
    int t = threadIdx.x, nt = blockDim.x;
    unsigned prefix = 0;
    int kk = (int)(NS * 0.2);   // 3276
    for (int bit = 31; bit >= 0; --bit) {
        unsigned cand = prefix | (1u << bit);
        unsigned mask = ~((1u << bit) - 1u);
        int local = 0;
        for (int i = t; i < NS; i += nt) {
            unsigned b = __float_as_uint(scores[i]);
            unsigned u = (b & 0x80000000u) ? ~b : (b | 0x80000000u);
            if ((u & mask) == cand) ++local;
        }
        if (t == 0) cnt = 0;
        __syncthreads();
        atomicAdd(&cnt, local);
        __syncthreads();
        int c = cnt;
        if (c >= kk) prefix = cand; else kk -= c;
        __syncthreads();
    }
    if (t == 0) {
        unsigned orig = (prefix & 0x80000000u) ? (prefix ^ 0x80000000u) : ~prefix;
        cutoff_out[0] = __uint_as_float(orig);
    }
}

// =====================================================================
// Segment construction (serial per batch-row, 128 independent threads)
// =====================================================================
__global__ void build_segments_kernel(const float* __restrict__ loss,
                                      const float* __restrict__ cutoff,
                                      int* __restrict__ segE, int* __restrict__ segCnt) {
    int b = threadIdx.x;
    if (b >= BSZ) return;
    float co = cutoff[0];
    int idxs[SEQ];
    int m = 0;
    for (int t = 0; t < SEQ; ++t)
        if (loss[(size_t)b * SEQ + t] > co) idxs[m++] = t;
    int cnt = 0;
    int* out = segE + b * MAXSEG;
    if (m > 0) {
        for (int i = 0; i < m - 1; ++i) {
            if (idxs[i + 1] - idxs[i] > MAL) {
                int cur = idxs[i];
                out[cnt++] = cur;
                while (cur < idxs[i + 1]) {
                    cur += MAL;
                    if (cur >= idxs[i + 1]) break;
                    out[cnt++] = cur;
                }
            } else {
                out[cnt++] = idxs[i];
            }
        }
        out[cnt++] = idxs[m - 1];
    }
    segCnt[b] = cnt;
}

__global__ void scan_offsets_kernel(const int* __restrict__ segCnt,
                                    int* __restrict__ segOff, int* __restrict__ total) {
    if (threadIdx.x == 0 && blockIdx.x == 0) {
        int s = 0;
        for (int b = 0; b < BSZ; ++b) { segOff[b] = s; s += segCnt[b]; }
        *total = s;
    }
}

__global__ void build_meta_kernel(const int* __restrict__ segE, const int* __restrict__ segCnt,
                                  const int* __restrict__ segOff,
                                  int* __restrict__ metaB, int* __restrict__ metaL,
                                  int* __restrict__ metaE, int Amax) {
    int b = threadIdx.x;
    if (b >= BSZ) return;
    int off = segOff[b];
    int last = 0;
    int cnt = segCnt[b];
    for (int j = 0; j < cnt; ++j) {
        int e = segE[b * MAXSEG + j];
        int a = off + j;
        if (a < Amax) { metaB[a] = b; metaL[a] = last; metaE[a] = e; }
        last = e;
    }
}

// =====================================================================
// Atom gather: imu_atoms / masks into d_out, transposed features into ws
// =====================================================================
__global__ void gather_atoms_kernel(const float* __restrict__ x, const float* __restrict__ bridge,
                                    const int* __restrict__ metaB, const int* __restrict__ metaL,
                                    const int* __restrict__ metaE, const int* __restrict__ total,
                                    int A,
                                    float* __restrict__ atomsOut, float* __restrict__ maskOut,
                                    float* __restrict__ featsOut) {
    int a = blockIdx.x;
    if (a >= *total || a >= A) return;
    int b = metaB[a], last = metaL[a], e = metaE[a];
    int le = e - last;
    int w  = le * 20;
    int ap = WTOT - w;
    for (int idx = threadIdx.x; idx < 6 * WTOT; idx += blockDim.x) {
        int c = idx / WTOT, col = idx - c * WTOT;
        int sc = col - ap;
        if (sc >= 0 && sc < w) {
            int t = last + sc / 20, p = sc % 20;
            atomsOut[(size_t)a * 7680 + idx] = x[(((size_t)b * SEQ + t) * 6 + c) * 20 + p];
            maskOut [(size_t)a * 7680 + idx] = 1.0f;
        }
    }
    int pad = MAL - le;
    for (int idx = threadIdx.x; idx < 640; idx += blockDim.x) {   // (D=64, MAL=10)
        int d = idx / MAL, r = idx - d * MAL;
        int i = r - pad;
        if (i >= 0 && i < le)
            featsOut[(size_t)a * 640 + idx] =
                bridge[((size_t)b * SEQ + last + i) * 64 + d];
    }
}

// =====================================================================
// Atom AE encoder (tiny convs -> VALU)
// =====================================================================
__global__ void ae1_kernel(const float* __restrict__ feats, const float* __restrict__ w,
                           const float* __restrict__ b, float* __restrict__ ha1, int A) {
    int tid = blockIdx.x * blockDim.x + threadIdx.x;
    if (tid >= A * 32) return;
    int a = tid >> 5, oc = tid & 31;
    const float* f = feats + (size_t)a * 640;
    float y[10];
    float bb = b[oc];
    for (int p = 0; p < 10; ++p) y[p] = bb;
    for (int ic = 0; ic < 64; ++ic) {
        const float* wv = w + (oc * 64 + ic) * 3;
        float w0 = wv[0], w1 = wv[1], w2 = wv[2];
        const float* row = f + ic * 10;
        for (int p = 0; p < 10; ++p) {
            float s = row[p] * w1;
            if (p > 0) s += row[p - 1] * w0;
            if (p < 9) s += row[p + 1] * w2;
            y[p] += s;
        }
    }
    float* out = ha1 + (size_t)a * 160 + oc * 5;
    for (int q = 0; q < 5; ++q) out[q] = fmaxf(fmaxf(y[2 * q], y[2 * q + 1]), 0.f);
}

__global__ void ae2_kernel(const float* __restrict__ ha1, const float* __restrict__ w,
                           const float* __restrict__ b, float* __restrict__ ha2, int A) {
    int tid = blockIdx.x * blockDim.x + threadIdx.x;
    if (tid >= A * 16) return;
    int a = tid >> 4, oc = tid & 15;
    const float* in = ha1 + (size_t)a * 160;
    float y[5];
    float bb = b[oc];
    for (int p = 0; p < 5; ++p) y[p] = bb;
    for (int ic = 0; ic < 32; ++ic) {
        const float* wv = w + (oc * 32 + ic) * 3;
        float w0 = wv[0], w1 = wv[1], w2 = wv[2];
        const float* row = in + ic * 5;
        for (int p = 0; p < 5; ++p) {
            float s = row[p] * w1;
            if (p > 0) s += row[p - 1] * w0;
            if (p < 4) s += row[p + 1] * w2;
            y[p] += s;
        }
    }
    float* out = ha2 + (size_t)a * 80 + oc * 5;
    for (int p = 0; p < 5; ++p) out[p] = fmaxf(y[p], 0.f);
}

__global__ void emb_kernel(const float* __restrict__ ha2, const float* __restrict__ w,
                           const float* __restrict__ b, _Float16* __restrict__ embh, int A) {
    int tid = blockIdx.x * blockDim.x + threadIdx.x;
    if (tid >= A * 32) return;
    int a = tid >> 5, j = tid & 31;
    float s = b[j];
    const float* in = ha2 + (size_t)a * 80;
    for (int i = 0; i < 80; ++i) s += in[i] * w[i * 32 + j];
    embh[(size_t)a * 32 + j] = (_Float16)s;
}

// =====================================================================
// Decoder: einsum(nci,cok)->relu->conv1d, fused through LDS (one sample / 32 lanes)
// =====================================================================
__global__ void decoder_kernel(const float* __restrict__ bridge,
                               const float* __restrict__ d1w, const float* __restrict__ d1b,
                               const float* __restrict__ d2w, const float* __restrict__ d2b,
                               float* __restrict__ imu_gen) {
    __shared__ float g[4][640];
    int lane = threadIdx.x & 31;
    int s    = threadIdx.x >> 5;             // 0..3
    int n    = blockIdx.x * 4 + s;
    const float* br = bridge + (size_t)n * 64;
    int o = lane;                            // output channel 0..31
    float acc[20];
    float bb = d1b[o];
#pragma unroll
    for (int ik = 0; ik < 20; ++ik) acc[ik] = bb;
    for (int c = 0; c < 16; ++c) {
        const float* wv = d1w + (c * 32 + o) * 5;
#pragma unroll
        for (int i = 0; i < 4; ++i) {
            float bv = br[c * 4 + i];
#pragma unroll
            for (int k = 0; k < 5; ++k) acc[i * 5 + k] += bv * wv[k];
        }
    }
#pragma unroll
    for (int ik = 0; ik < 20; ++ik) g[s][o * 20 + ik] = fmaxf(acc[ik], 0.f);
    __syncthreads();
    for (int idx = lane; idx < 120; idx += 32) {
        int oc = idx / 20, p = idx - oc * 20;
        float y = d2b[oc];
        for (int ic = 0; ic < 32; ++ic) {
            const float* wv = d2w + (oc * 32 + ic) * 3;
            const float* gg = g[s] + ic * 20;
            float v = gg[p] * wv[1];
            if (p > 0)  v += gg[p - 1] * wv[0];
            if (p < 19) v += gg[p + 1] * wv[2];
            y += v;
        }
        imu_gen[(size_t)n * 120 + idx] = y;
    }
}

// =====================================================================
// Utility kernels
// =====================================================================
__global__ void zero_f32_kernel(float* p, size_t n) {
    for (size_t i = blockIdx.x * (size_t)blockDim.x + threadIdx.x; i < n;
         i += (size_t)gridDim.x * blockDim.x) p[i] = 0.f;
}
__global__ void zero_f16_kernel(_Float16* p, size_t n) {
    for (size_t i = blockIdx.x * (size_t)blockDim.x + threadIdx.x; i < n;
         i += (size_t)gridDim.x * blockDim.x) p[i] = (_Float16)0.f;
}
__global__ void copy_f32_kernel(float* __restrict__ dst, const float* __restrict__ src, size_t n) {
    for (size_t i = blockIdx.x * (size_t)blockDim.x + threadIdx.x; i < n;
         i += (size_t)gridDim.x * blockDim.x) dst[i] = src[i];
}

// =====================================================================
// Host launcher
// =====================================================================
extern "C" void kernel_launch(void* const* d_in, const int* in_sizes, int n_in,
                              void* d_out, int out_size, void* d_ws, size_t ws_size,
                              hipStream_t stream) {
    (void)in_sizes; (void)n_in; (void)ws_size;

    const float* x        = (const float*)d_in[0];
    const float* imu_mask = (const float*)d_in[1];
    // d_in[2] = imu_len (unused by the reference computation)
    const float* c1w = (const float*)d_in[3];  const float* c1b = (const float*)d_in[4];
    const float* c2w = (const float*)d_in[5];  const float* c2b = (const float*)d_in[6];
    const float* c3w = (const float*)d_in[7];  const float* c3b = (const float*)d_in[8];
    const float* brw = (const float*)d_in[9];  const float* brb = (const float*)d_in[10];
    const float* f1w = (const float*)d_in[11]; const float* f1b = (const float*)d_in[12];
    const float* f2w = (const float*)d_in[13]; const float* f2b = (const float*)d_in[14];
    const float* d1w = (const float*)d_in[15]; const float* d1b = (const float*)d_in[16];
    const float* d2w = (const float*)d_in[17]; const float* d2b = (const float*)d_in[18];
    const float* ae1w = (const float*)d_in[19]; const float* ae1b = (const float*)d_in[20];
    const float* ae2w = (const float*)d_in[21]; const float* ae2b = (const float*)d_in[22];
    const float* ae3w = (const float*)d_in[23]; const float* ae3b = (const float*)d_in[24];
    const float* ad1w = (const float*)d_in[25]; const float* ad1b = (const float*)d_in[26];
    const float* ad2w = (const float*)d_in[27]; const float* ad2b = (const float*)d_in[28];

    // ---- output layout (floats), A recovered from out_size ----
    const size_t fixed = 1966080ull + 4ull * 1048576ull + 16384ull;  // 6,176,768
    long long rem = (long long)out_size - (long long)fixed;
    int A = (rem > 0) ? (int)(rem / (3ll * 7680)) : 0;
    if (A < 1) A = 1;
    int Apad   = (A + 15) & ~15;
    int Atiles = Apad >> 4;

    float* outF = (float*)d_out;
    size_t o_imu_gen     = 0;
    size_t o_atom_gen    = 1966080ull;
    size_t o_mask        = o_atom_gen + (size_t)A * 7680;
    size_t o_atoms       = o_mask     + (size_t)A * 7680;
    size_t o_bridge_resh = o_atoms    + (size_t)A * 7680;
    size_t o_forcast_in  = o_bridge_resh + 1048576ull;
    size_t o_forcast     = o_forcast_in  + 1048576ull;
    size_t o_fmask       = o_forcast     + 1048576ull;
    size_t o_loss        = o_fmask       + 1048576ull;

    // ---- workspace carving (256B aligned) ----
    char* base = (char*)d_ws;
    size_t off = 0;
    auto carve = [&](size_t bytes) -> void* {
        void* p = base + off;
        off += (bytes + 255) & ~(size_t)255;
        return p;
    };
    float*    h1    = (float*)carve((size_t)NS * 160 * 4);
    float*    h2    = (float*)carve((size_t)NS * 160 * 4);
    _Float16* cnn   = (_Float16*)carve((size_t)NS * 64 * 2);
    _Float16* shft  = (_Float16*)carve((size_t)NS * 64 * 2);
    _Float16* feat  = (_Float16*)carve((size_t)NS * 256 * 2);
    _Float16* pBr   = (_Float16*)carve((size_t)64 * 64 * 2);
    _Float16* pF1   = (_Float16*)carve((size_t)64 * 256 * 2);
    _Float16* pF2   = (_Float16*)carve((size_t)256 * 64 * 2);
    _Float16* pAd1  = (_Float16*)carve((size_t)32 * 256 * 2);
    _Float16* pAd2  = (_Float16*)carve((size_t)256 * 7680 * 2);
    float*    scores  = (float*)carve((size_t)NS * 4);
    float*    cutoff  = (float*)carve(256);
    int*      segE    = (int*)carve((size_t)BSZ * MAXSEG * 4);
    int*      segCnt  = (int*)carve(BSZ * 4);
    int*      segOff  = (int*)carve(BSZ * 4);
    int*      segTot  = (int*)carve(256);
    int*      metaB   = (int*)carve((size_t)A * 4);
    int*      metaL   = (int*)carve((size_t)A * 4);
    int*      metaE   = (int*)carve((size_t)A * 4);
    float*    afeat   = (float*)carve((size_t)A * 640 * 4);
    float*    ha1     = (float*)carve((size_t)A * 160 * 4);
    float*    ha2     = (float*)carve((size_t)A * 80 * 4);
    _Float16* embh    = (_Float16*)carve((size_t)Apad * 32 * 2);
    _Float16* hdh     = (_Float16*)carve((size_t)Apad * 256 * 2);

    // ---- 1) pack weights into WMMA B-fragment layout ----
    pack_b_kernel<<<64,   256, 0, stream>>>(brw,  64,  64,   pBr);
    pack_b_kernel<<<64,   256, 0, stream>>>(f1w,  64,  256,  pF1);
    pack_b_kernel<<<64,   256, 0, stream>>>(f2w,  256, 64,   pF2);
    pack_b_kernel<<<64,   256, 0, stream>>>(ad1w, 32,  256,  pAd1);
    pack_b_kernel<<<2048, 256, 0, stream>>>(ad2w, 256, 7680, pAd2);

    // ---- 2) encoder conv stack ----
    conv1_kernel<<<(NS * 16 + 255) / 256, 256, 0, stream>>>(x,  c1w, c1b, h1);
    conv2_kernel<<<(NS * 32 + 255) / 256, 256, 0, stream>>>(h1, c2w, c2b, h2);
    conv3_kernel<<<(NS * 32 + 255) / 256, 256, 0, stream>>>(h2, c3w, c3b, cnn);

    // ---- 3) bridge GEMM (sigmoid), f32 -> forcast_in region, f16 shifted -> shft ----
    zero_f16_kernel<<<1024, 256, 0, stream>>>(shft, (size_t)NS * 64);
    gemm_wmma<2><<<(1024 + 7) / 8, 256, 0, stream>>>(
        cnn, 64, pBr, brb, 1024, 4,
        outF + o_forcast_in, 64, NS, shft, 64, 2, 1);
    copy_f32_kernel<<<1024, 256, 0, stream>>>(outF + o_bridge_resh, outF + o_forcast_in,
                                              (size_t)NS * 64);

    // ---- 4) forecast MLP: f1 (relu, f16 out) then f2 ----
    gemm_wmma<2><<<(1024 + 7) / 8, 256, 0, stream>>>(
        shft, 64, pF1, f1b, 1024, 16,
        (float*)nullptr, 0, NS, feat, 256, 1, 0);
    gemm_wmma<8><<<(1024 + 7) / 8, 256, 0, stream>>>(
        feat, 256, pF2, f2b, 1024, 4,
        outF + o_forcast, 64, NS, (_Float16*)nullptr, 0, 0, 0);

    // ---- 5) loss / fmask / scores ----
    loss_kernel<<<(NS + 255) / 256, 256, 0, stream>>>(
        outF + o_forcast, outF + o_forcast_in, imu_mask,
        outF + o_fmask, outF + o_loss, scores);

    // ---- 6) decoder -> imu_gen ----
    decoder_kernel<<<NS / 4, 128, 0, stream>>>(
        outF + o_forcast_in, d1w, d1b, d2w, d2b, outF + o_imu_gen);

    // ---- 7) cutoff + segments + meta ----
    select_cutoff_kernel<<<1, 256, 0, stream>>>(scores, cutoff);
    build_segments_kernel<<<1, 128, 0, stream>>>(outF + o_loss, cutoff, segE, segCnt);
    scan_offsets_kernel<<<1, 1, 0, stream>>>(segCnt, segOff, segTot);
    build_meta_kernel<<<1, 128, 0, stream>>>(segE, segCnt, segOff, metaB, metaL, metaE, A);

    // ---- 8) atom gather (zero-init first: padded regions stay zero) ----
    zero_f32_kernel<<<4096, 256, 0, stream>>>(outF + o_atom_gen, 3ull * (size_t)A * 7680);
    zero_f32_kernel<<<1024, 256, 0, stream>>>(afeat, (size_t)A * 640);
    gather_atoms_kernel<<<A, 128, 0, stream>>>(
        x, outF + o_forcast_in, metaB, metaL, metaE, segTot, A,
        outF + o_atoms, outF + o_mask, afeat);

    // ---- 9) atom AE encoder + embedding ----
    ae1_kernel<<<(A * 32 + 255) / 256, 256, 0, stream>>>(afeat, ae1w, ae1b, ha1, A);
    ae2_kernel<<<(A * 16 + 255) / 256, 256, 0, stream>>>(ha1,  ae2w, ae2b, ha2, A);
    emb_kernel<<<(A * 32 + 255) / 256, 256, 0, stream>>>(ha2,  ae3w, ae3b, embh, A);

    // ---- 10) atom decoder GEMMs (WMMA) ----
    gemm_wmma<1><<<(Atiles + 7) / 8, 256, 0, stream>>>(
        embh, 32, pAd1, ad1b, Atiles, 16,
        (float*)nullptr, 0, A, hdh, 256, 1, 0);
    gemm_wmma<8><<<(Atiles + 7) / 8, 256, 0, stream>>>(
        hdh, 256, pAd2, ad2b, Atiles, 480,
        outF + o_atom_gen, 7680, A, (_Float16*)nullptr, 0, 0, 0);
}
